// HomoGNN_83056077570947
// MI455X (gfx1250) — matrix operand order repaired and checked by reading, not compiled, not run on hardware
//
#include <hip/hip_runtime.h>
#include <hip/hip_bf16.h>
#include <math.h>

typedef __attribute__((ext_vector_type(16))) _Float16 v16h;
typedef __attribute__((ext_vector_type(8)))  _Float16 v8h;
typedef __attribute__((ext_vector_type(8)))  float    v8f;
typedef int v4i_ls __attribute__((vector_size(4 * sizeof(int))));  // matches builtin param type

#define DDIM 256
#define HEADS 4
#define NEG_ATT 0.2f
#define NEG_ACT 0.01f
#define LDS_PAD 8   // halves of padding per LDS row (bank-conflict avoidance)

#if defined(__has_builtin)
#if __has_builtin(__builtin_amdgcn_global_load_async_to_lds_b128) && \
    __has_builtin(__builtin_amdgcn_s_wait_asynccnt)
#define USE_ASYNC_LDS 1
#endif
#endif

// ---------------------------------------------------------------- utilities

__device__ __forceinline__ float wave_sum32(float v) {
#pragma unroll
  for (int off = 16; off > 0; off >>= 1) v += __shfl_xor(v, off, 32);
  return v;
}

__device__ __forceinline__ void atomic_max_float(float* addr, float val) {
  if (val >= 0.f) {
    atomicMax((int*)addr, __float_as_int(val));
  } else {
    atomicMin((unsigned int*)addr, __float_as_uint(val));
  }
}

// ------------------------------------------------------- f16 pre-conversion

__global__ void k_f32_to_f16(const float* __restrict__ src,
                             _Float16* __restrict__ dst, size_t count) {
  const size_t i = (size_t)blockIdx.x * blockDim.x + threadIdx.x;
  if (i < count) dst[i] = (_Float16)src[i];
}

// W[K][256] (f32) -> Wt[256][K] (f16, transposed so a B-fragment column is contiguous)
__global__ void k_w_to_f16t(const float* __restrict__ W,
                            _Float16* __restrict__ Wt, int K) {
  const int idx = blockIdx.x * blockDim.x + threadIdx.x;
  if (idx >= K * DDIM) return;
  const int k = idx >> 8, c = idx & (DDIM - 1);
  Wt[(size_t)c * K + k] = (_Float16)W[idx];
}

// ---------------------------------------------------------------- WMMA GEMM
// Y[n,256] = Xh[n,K] @ W[K,256] + bias, with Wt = W^T in f16.
// Block = 256 threads = 8 waves; block tile 128(M) x 64(N); grid.y = 4.
// B tile (64 cols x K) staged in LDS via async-to-LDS b128; fragments via ds_load_b128.
__global__ __launch_bounds__(256)
void k_gemm_wmma(const _Float16* __restrict__ Xh, const _Float16* __restrict__ Wt,
                 const float* __restrict__ bias, float* __restrict__ Y,
                 int n, int K) {
  __shared__ _Float16 Wt_s[64 * (DDIM + LDS_PAD)];   // max 64*264*2 = 33792 B

  const int lane = threadIdx.x & 31;
  const int wave = threadIdx.x >> 5;
  const int row0 = blockIdx.x * 128 + wave * 16;
  const int n0   = blockIdx.y * 64;
  const int l    = lane & 15;
  const int hi   = lane >> 4;            // lane group 0/1
  const int ldsK = K + LDS_PAD;          // padded LDS row stride (halves)

  // ---- stage Wt[n0 .. n0+63][0..K) into LDS (16B chunks) ----
  {
    const int sh = (K == 128) ? 4 : 5;         // chunks per column = K/8
    const int nChunks = 64 << sh;
    for (int c = threadIdx.x; c < nChunks; c += 256) {
      const int colL = c >> sh;
      const int kc   = (c & ((1 << sh) - 1)) << 3;
      const _Float16* gp = Wt + (size_t)(n0 + colL) * K + kc;
      _Float16*       lp = &Wt_s[colL * ldsK + kc];
#if defined(USE_ASYNC_LDS)
      __builtin_amdgcn_global_load_async_to_lds_b128(
          (v4i_ls*)gp, (v4i_ls*)lp, 0, 0);
#else
      *(v8h*)lp = *(const v8h*)gp;
#endif
    }
  }
#if defined(USE_ASYNC_LDS)
  __builtin_amdgcn_s_wait_asynccnt(0);
#endif
  __syncthreads();

  int m = row0 + l;
  if (m >= n) m = n - 1;   // clamp: WMMA needs EXEC all-ones; store is masked
  const _Float16* __restrict__ arow = Xh + (size_t)m * K;

  v8f acc[4] = {};

  for (int k0 = 0; k0 < K; k0 += 32) {
    // A fragment (16x32 f16): lane<16 -> K = k0+0..7, k0+16..23 ; lane>=16 -> +8
    const v8h a_lo = *(const v8h*)(arow + k0 + hi * 8);
    const v8h a_hi = *(const v8h*)(arow + k0 + 16 + hi * 8);
    v16h a;
#pragma unroll
    for (int j = 0; j < 8; ++j) { a[j] = a_lo[j]; a[8 + j] = a_hi[j]; }

#pragma unroll
    for (int t = 0; t < 4; ++t) {
      // B fragment (32x16 f16, column-major per lane):
      // lane<16: col n0+t*16+l, K = k0+0..7, k0+16..23 ; lane>=16 -> +8
      const _Float16* bp = &Wt_s[(t * 16 + l) * ldsK + k0 + hi * 8];
      const v8h b_lo = *(const v8h*)bp;
      const v8h b_hi = *(const v8h*)(bp + 16);
      v16h b;
#pragma unroll
      for (int j = 0; j < 8; ++j) { b[j] = b_lo[j]; b[8 + j] = b_hi[j]; }
      acc[t] = __builtin_amdgcn_wmma_f32_16x16x32_f16(
          false, a, false, b, (short)0, acc[t], false, false);
    }
  }

  // C/D layout: VGPR i -> row = row0 + i + 8*hi, col = n0 + t*16 + l
#pragma unroll
  for (int t = 0; t < 4; ++t) {
    const int col = n0 + t * 16 + l;
    const float bv = bias[col];
#pragma unroll
    for (int i = 0; i < 8; ++i) {
      const int row = row0 + i + 8 * hi;
      if (row < n) Y[(size_t)row * DDIM + col] = acc[t][i] + bv;
    }
  }
}

// ------------------------------------------------------- init / edge kernels

// out[r][c] = bias[c]; m = -inf; z = 0
__global__ void k_init(float* __restrict__ out, const float* __restrict__ bias,
                       float* __restrict__ mbuf, float* __restrict__ zbuf, int n) {
  size_t idx = (size_t)blockIdx.x * blockDim.x + threadIdx.x;
  if (idx < (size_t)n * DDIM) out[idx] = bias[idx & (DDIM - 1)];
  if (idx < (size_t)n * HEADS) {
    mbuf[idx] = -3.402823466e38f;
    zbuf[idx] = 0.f;
  }
}

// one wave per edge: logits[e][h] = sum_c att[h][c]*lrelu(xl[s]+xr[d]); atomicMax m[d][h]
__global__ __launch_bounds__(256)
void k_edge_logits(const float* __restrict__ xl, const float* __restrict__ xr,
                   const int* __restrict__ eidx, const float* __restrict__ att,
                   float* __restrict__ logits, float* __restrict__ mbuf,
                   int E, int ET) {
  const int gw = (int)(((size_t)blockIdx.x * blockDim.x + threadIdx.x) >> 5);
  const int lane = threadIdx.x & 31;
  if (gw >= ET) return;
  int s, d;
  if (gw < E) { s = eidx[gw]; d = eidx[E + gw]; }
  else        { s = gw - E;   d = gw - E; }

  const float* __restrict__ pl = xl + (size_t)s * DDIM;
  const float* __restrict__ pr = xr + (size_t)d * DDIM;
  float hs[HEADS] = {0.f, 0.f, 0.f, 0.f};
#pragma unroll
  for (int i = 0; i < 8; ++i) {
    const int dim = i * 32 + lane;           // head = i>>1 for all 32 lanes
    float v = pl[dim] + pr[dim];
    v = v > 0.f ? v : NEG_ATT * v;
    hs[i >> 1] += v * att[dim];
  }
#pragma unroll
  for (int h = 0; h < HEADS; ++h) hs[h] = wave_sum32(hs[h]);
  if (lane == 0) {
#pragma unroll
    for (int h = 0; h < HEADS; ++h) {
      logits[(size_t)gw * HEADS + h] = hs[h];
      atomic_max_float(&mbuf[(size_t)d * HEADS + h], hs[h]);
    }
  }
}

// thread per (e,h): a = exp(logit - m[dst]); store in place; z[dst] += a
__global__ void k_edge_expsum(const int* __restrict__ eidx, float* __restrict__ ab,
                              const float* __restrict__ mbuf, float* __restrict__ zbuf,
                              int E, int ET) {
  const size_t t = (size_t)blockIdx.x * blockDim.x + threadIdx.x;
  if (t >= (size_t)ET * HEADS) return;
  const int e = (int)(t >> 2), h = (int)(t & 3);
  const int d = (e < E) ? eidx[E + e] : e - E;
  const float a = expf(ab[t] - mbuf[(size_t)d * HEADS + h]);
  ab[t] = a;
  atomicAdd(&zbuf[(size_t)d * HEADS + h], a);
}

// one wave per edge: out[dst] += (a/z[dst])[h(dim)] * xl[src]
__global__ __launch_bounds__(256)
void k_edge_aggregate(const float* __restrict__ xl, const int* __restrict__ eidx,
                      const float* __restrict__ ab, const float* __restrict__ zbuf,
                      float* __restrict__ out, int E, int ET) {
  const int gw = (int)(((size_t)blockIdx.x * blockDim.x + threadIdx.x) >> 5);
  const int lane = threadIdx.x & 31;
  if (gw >= ET) return;
  int s, d;
  if (gw < E) { s = eidx[gw]; d = eidx[E + gw]; }
  else        { s = gw - E;   d = gw - E; }

  float alpha[HEADS];
#pragma unroll
  for (int h = 0; h < HEADS; ++h)
    alpha[h] = ab[(size_t)gw * HEADS + h] / zbuf[(size_t)d * HEADS + h];

  const float* __restrict__ pl = xl + (size_t)s * DDIM;
  float* __restrict__ po = out + (size_t)d * DDIM;
#pragma unroll
  for (int i = 0; i < 8; ++i) {
    const int dim = i * 32 + lane;
    atomicAdd(&po[dim], alpha[i >> 1] * pl[dim]);
  }
}

// ---------------------------------------------------------------- batch norm

__global__ void k_zero_stats(float* __restrict__ st) { st[threadIdx.x] = 0.f; }

__global__ __launch_bounds__(256)
void k_bn_stats(const float* __restrict__ h, float* __restrict__ sum,
                float* __restrict__ sumsq, int n) {
  const int col = threadIdx.x;
  float s = 0.f, s2 = 0.f;
  for (int r = blockIdx.x; r < n; r += gridDim.x) {
    const float v = h[(size_t)r * DDIM + col];
    s += v;
    s2 += v * v;
  }
  atomicAdd(&sum[col], s);
  atomicAdd(&sumsq[col], s2);
}

__global__ void k_bn_apply(float* __restrict__ h, const float* __restrict__ sum,
                           const float* __restrict__ sumsq,
                           const float* __restrict__ gamma,
                           const float* __restrict__ beta, int n) {
  const size_t idx = (size_t)blockIdx.x * blockDim.x + threadIdx.x;
  if (idx >= (size_t)n * DDIM) return;
  const int col = (int)(idx & (DDIM - 1));
  const float inv_n = 1.f / (float)n;
  const float mu = sum[col] * inv_n;
  const float var = sumsq[col] * inv_n - mu * mu;   // biased var (matches jnp)
  float v = (h[idx] - mu) * rsqrtf(var + 1e-5f) * gamma[col] + beta[col];
  h[idx] = v > 0.f ? v : NEG_ACT * v;
}

// --------------------------------------------------------------- classifier

__global__ __launch_bounds__(256)
void k_classifier(const float* __restrict__ h, const float* __restrict__ Wc,
                  const float* __restrict__ bc, float* __restrict__ out, int n) {
  const int gw = (int)(((size_t)blockIdx.x * blockDim.x + threadIdx.x) >> 5);
  const int lane = threadIdx.x & 31;
  if (gw >= n) return;
  const float* __restrict__ hr = h + (size_t)gw * DDIM;
  float c0 = 0.f, c1 = 0.f;
#pragma unroll
  for (int i = 0; i < 8; ++i) {
    const int dim = i * 32 + lane;
    const float v = hr[dim];
    c0 += v * Wc[dim * 2];
    c1 += v * Wc[dim * 2 + 1];
  }
  c0 = wave_sum32(c0);
  c1 = wave_sum32(c1);
  if (lane == 0) {
    out[(size_t)gw * 2 + 0] = c0 + bc[0];
    out[(size_t)gw * 2 + 1] = c1 + bc[1];
  }
}

// -------------------------------------------------------------------- driver

extern "C" void kernel_launch(void* const* d_in, const int* in_sizes, int n_in,
                              void* d_out, int out_size, void* d_ws, size_t ws_size,
                              hipStream_t stream) {
  const float* x    = (const float*)d_in[0];
  const int*   eidx = (const int*)d_in[1];
  const float* Wl1  = (const float*)d_in[2];
  const float* bl1  = (const float*)d_in[3];
  const float* Wr1  = (const float*)d_in[4];
  const float* br1  = (const float*)d_in[5];
  const float* att1 = (const float*)d_in[6];
  const float* b1   = (const float*)d_in[7];
  const float* Wl2  = (const float*)d_in[8];
  const float* bl2  = (const float*)d_in[9];
  const float* Wr2  = (const float*)d_in[10];
  const float* br2  = (const float*)d_in[11];
  const float* att2 = (const float*)d_in[12];
  const float* b2   = (const float*)d_in[13];
  const float* gam  = (const float*)d_in[14];
  const float* bet  = (const float*)d_in[15];
  const float* Wc   = (const float*)d_in[16];
  const float* bc   = (const float*)d_in[17];

  const int N  = in_sizes[0] / 128;   // F_IN = 128
  const int E  = in_sizes[1] / 2;     // edge_index is [2, E]
  const int ET = E + N;               // + self loops

  // workspace layout (f32 region, then f16 region; all 16B aligned)
  float* xl = (float*)d_ws;                      // N*256
  float* xr = xl + (size_t)N * DDIM;             // N*256
  float* hb = xr + (size_t)N * DDIM;             // N*256
  float* ab = hb + (size_t)N * DDIM;             // ET*4
  float* mb = ab + (size_t)ET * HEADS;           // N*4
  float* zb = mb + (size_t)N * HEADS;            // N*4
  float* st = zb + (size_t)N * HEADS;            // 512
  _Float16* Xh = (_Float16*)(st + 512);          // N*256 halves (max)
  _Float16* Wt = Xh + (size_t)N * DDIM;          // 256*256 halves

  const dim3 gG((N + 127) / 128, 4), tG(256);
  const int nBlk = (int)(((size_t)N * DDIM + 255) / 256);
  const int eWav = (ET + 7) / 8;
  const int eThr = (int)(((size_t)ET * HEADS + 255) / 256);
  const int wBlk1 = (128 * DDIM + 255) / 256, wBlk2 = (256 * DDIM + 255) / 256;

  // ---- layer 1 (K = 128) ----
  k_f32_to_f16<<<(int)(((size_t)N * 128 + 255) / 256), 256, 0, stream>>>(x, Xh, (size_t)N * 128);
  k_w_to_f16t<<<wBlk1, 256, 0, stream>>>(Wl1, Wt, 128);
  k_gemm_wmma<<<gG, tG, 0, stream>>>(Xh, Wt, bl1, xl, N, 128);
  k_w_to_f16t<<<wBlk1, 256, 0, stream>>>(Wr1, Wt, 128);
  k_gemm_wmma<<<gG, tG, 0, stream>>>(Xh, Wt, br1, xr, N, 128);
  k_init<<<nBlk, 256, 0, stream>>>(hb, b1, mb, zb, N);
  k_edge_logits<<<eWav, 256, 0, stream>>>(xl, xr, eidx, att1, ab, mb, E, ET);
  k_edge_expsum<<<eThr, 256, 0, stream>>>(eidx, ab, mb, zb, E, ET);
  k_edge_aggregate<<<eWav, 256, 0, stream>>>(xl, eidx, ab, zb, hb, E, ET);
  k_zero_stats<<<1, 512, 0, stream>>>(st);
  k_bn_stats<<<512, 256, 0, stream>>>(hb, st, st + 256, N);
  k_bn_apply<<<nBlk, 256, 0, stream>>>(hb, st, st + 256, gam, bet, N);

  // ---- layer 2 (K = 256) ----
  k_f32_to_f16<<<nBlk, 256, 0, stream>>>(hb, Xh, (size_t)N * DDIM);
  k_w_to_f16t<<<wBlk2, 256, 0, stream>>>(Wl2, Wt, 256);
  k_gemm_wmma<<<gG, tG, 0, stream>>>(Xh, Wt, bl2, xl, N, 256);
  k_w_to_f16t<<<wBlk2, 256, 0, stream>>>(Wr2, Wt, 256);
  k_gemm_wmma<<<gG, tG, 0, stream>>>(Xh, Wt, br2, xr, N, 256);
  k_init<<<nBlk, 256, 0, stream>>>(hb, b2, mb, zb, N);
  k_edge_logits<<<eWav, 256, 0, stream>>>(xl, xr, eidx, att2, ab, mb, E, ET);
  k_edge_expsum<<<eThr, 256, 0, stream>>>(eidx, ab, mb, zb, E, ET);
  k_edge_aggregate<<<eWav, 256, 0, stream>>>(xl, eidx, ab, zb, hb, E, ET);
  k_zero_stats<<<1, 512, 0, stream>>>(st);
  k_bn_stats<<<512, 256, 0, stream>>>(hb, st, st + 256, N);
  k_bn_apply<<<nBlk, 256, 0, stream>>>(hb, st, st + 256, gam, bet, N);

  // ---- classifier ----
  k_classifier<<<(N + 7) / 8, 256, 0, stream>>>(hb, Wc, bc, (float*)d_out, N);
}
